// Decoder_18545668784633
// MI455X (gfx1250) — matrix operand order repaired and checked by reading, not compiled, not run on hardware
//
#include <hip/hip_runtime.h>
#include <hip/hip_bf16.h>

typedef __attribute__((ext_vector_type(16))) __bf16 v16bf;
typedef __attribute__((ext_vector_type(8)))  float  v8f;

// Problem constants (from reference)
constexpr int Bsz = 32;
constexpr int T   = 512;
constexpr int D   = 1024;
constexpr int U   = 1024;
constexpr int N3  = 3 * U;            // 3072 gate columns
constexpr int KT  = D / 32;           // 32 k-tiles of K=32
constexpr int NT_GATE = U / 16;       // 64 n-tiles per gate
constexpr int NT_ALL  = N3 / 16;      // 192 n-tiles total
constexpr int FRAG_ELEMS = 512;       // 32 lanes * 16 bf16 per fragment
constexpr int AFRAGS    = 2 * KT;     // 64 A-fragments per 32x1024 operand
constexpr int AOP_ELEMS = AFRAGS * FRAG_ELEMS;   // 32768 bf16 per operand

// ---------------------------------------------------------------------------
// Weight swizzle: f32 row-major [K=1024][N=3072] -> bf16 WMMA-B fragments.
// Fragment f = nt*KT + kt holds the 32x16 tile (k0=kt*32, n0=nt*16).
// Lane L stores n = n0+(L&15), k = k0+8*(L>>4)+{0..7,16..23} (16 bf16 = 32 bytes, lane-contiguous).
// ---------------------------------------------------------------------------
__global__ void swizzle_weights(const float* __restrict__ W,
                                __bf16* __restrict__ out) {
  int tid  = blockIdx.x * blockDim.x + threadIdx.x;   // frag*32 + lane
  int lane = tid & 31;
  int f    = tid >> 5;                 // nt*KT + kt
  int kt   = f & (KT - 1);
  int nt   = f >> 5;                   // KT == 32
  int k    = kt * 32 + ((lane >> 4) << 3);
  int n0   = nt * 16 + (lane & 15);
  const float* src = W + (size_t)k * N3 + n0;
  __bf16* dst = out + (size_t)f * FRAG_ELEMS + lane * 16;
#pragma unroll
  for (int j = 0; j < 16; ++j) dst[j] = (__bf16)src[(size_t)(j + ((j >> 3) << 3)) * N3];
}

// ---------------------------------------------------------------------------
// Pack a 32x1024 f32 operand into 64 bf16 WMMA A-fragments (16x32 each).
// ISA layout (16-bit A): lanes 0-15 -> rows m, K = {0..7, 16..23};
//                        lanes 16-31 -> rows m, K = {8..15, 24..31}.
// blockIdx.y selects the operand instance (timestep for x).
// ---------------------------------------------------------------------------
__global__ void pack_a_frags(const float* __restrict__ Abase, long rowStride,
                             long opSrcOff, __bf16* __restrict__ outBase) {
  const float* A = Abase + (size_t)blockIdx.y * opSrcOff;
  __bf16* out = outBase + (size_t)blockIdx.y * AOP_ELEMS;
  int tid  = blockIdx.x * blockDim.x + threadIdx.x;   // frag*32 + lane
  int lane = tid & 31;
  int f    = tid >> 5;                 // mt*KT + kt
  int kt   = f & (KT - 1);
  int mt   = f >> 5;
  int m    = mt * 16 + (lane & 15);
  int kb   = kt * 32 + ((lane >> 4) << 3);    // +0 or +8
  const float* r = A + (long)m * rowStride + kb;
  float4 a0 = *(const float4*)(r);
  float4 a1 = *(const float4*)(r + 4);
  float4 a2 = *(const float4*)(r + 16);
  float4 a3 = *(const float4*)(r + 20);
  v16bf fr;
  fr[0]  = (__bf16)a0.x; fr[1]  = (__bf16)a0.y; fr[2]  = (__bf16)a0.z; fr[3]  = (__bf16)a0.w;
  fr[4]  = (__bf16)a1.x; fr[5]  = (__bf16)a1.y; fr[6]  = (__bf16)a1.z; fr[7]  = (__bf16)a1.w;
  fr[8]  = (__bf16)a2.x; fr[9]  = (__bf16)a2.y; fr[10] = (__bf16)a2.z; fr[11] = (__bf16)a2.w;
  fr[12] = (__bf16)a3.x; fr[13] = (__bf16)a3.y; fr[14] = (__bf16)a3.z; fr[15] = (__bf16)a3.w;
  *(v16bf*)(out + (size_t)f * FRAG_ELEMS + lane * 16) = fr;
}

__device__ __forceinline__ v8f wmma_bf16(v16bf a, v16bf b, v8f c) {
  return __builtin_amdgcn_wmma_f32_16x16x32_bf16(
      /*neg_a=*/false, a, /*neg_b=*/false, b,
      /*c_mod=*/(short)0, c, /*reuse_a=*/false, /*reuse_b=*/false);
}

// ---------------------------------------------------------------------------
// One fused GRU layer step:  g = A*W + H*U (+bi,+br), gated update of H.
// All GEMM operands pre-packed bf16 fragments; inner loop = pure vmem + wmma.
// One wave per (mt, nc): 16 batch rows x 16 u-columns, all 3 gates.
// Epilogue writes h_new as f32 AND as bf16 A-fragments for the next consumer.
// ---------------------------------------------------------------------------
__global__ void __launch_bounds__(32)
gru_step(const __bf16* __restrict__ Afrag,          // input operand fragments
         const float*  __restrict__ H,              // h state, f32 (read-only)
         const __bf16* __restrict__ Hfrag,          // h state fragments
         const __bf16* __restrict__ Wf, const __bf16* __restrict__ Uf,
         const float* __restrict__ bi, const float* __restrict__ br,
         float*  __restrict__ Hout,                 // h_next f32
         __bf16* __restrict__ Hfragout,             // h_next fragments
         float*  __restrict__ Yout, long Ystride) {
  const int lane = threadIdx.x;        // 0..31
  const int gw   = blockIdx.x;         // 0..127
  const int mt   = gw & 1;             // batch-row tile (rows mt*16..)
  const int nc   = gw >> 1;            // u-column tile (cols nc*16..)

  const int ntz = 0 * NT_GATE + nc;
  const int ntr = 1 * NT_GATE + nc;
  const int nth = 2 * NT_GATE + nc;

  v8f az  = {};   // z gate: x-side + h-side merged
  v8f ar  = {};   // r gate: merged
  v8f axh = {};   // h gate, input (W) side   (tanh(xh + r*hh) needs them apart)
  v8f ahh = {};   // h gate, recurrent (U) side

  const __bf16* af = Afrag + (size_t)(mt * KT) * FRAG_ELEMS + lane * 16;
  const __bf16* hf = Hfrag + (size_t)(mt * KT) * FRAG_ELEMS + lane * 16;
  const __bf16* wz = Wf + (size_t)(ntz * KT) * FRAG_ELEMS + lane * 16;
  const __bf16* wr = Wf + (size_t)(ntr * KT) * FRAG_ELEMS + lane * 16;
  const __bf16* wh = Wf + (size_t)(nth * KT) * FRAG_ELEMS + lane * 16;
  const __bf16* uz = Uf + (size_t)(ntz * KT) * FRAG_ELEMS + lane * 16;
  const __bf16* ur = Uf + (size_t)(ntr * KT) * FRAG_ELEMS + lane * 16;
  const __bf16* uh = Uf + (size_t)(nth * KT) * FRAG_ELEMS + lane * 16;

  for (int kt = 0; kt < KT; ++kt) {
    v16bf a1 = *(const v16bf*)(af);                 // input operand fragment
    v16bf a2 = *(const v16bf*)(hf);                 // recurrent operand fragment
    v16bf bz1 = *(const v16bf*)(wz);  az  = wmma_bf16(a1, bz1, az);
    v16bf br1 = *(const v16bf*)(wr);  ar  = wmma_bf16(a1, br1, ar);
    v16bf bh1 = *(const v16bf*)(wh);  axh = wmma_bf16(a1, bh1, axh);
    v16bf bz2 = *(const v16bf*)(uz);  az  = wmma_bf16(a2, bz2, az);
    v16bf br2 = *(const v16bf*)(ur);  ar  = wmma_bf16(a2, br2, ar);
    v16bf bh2 = *(const v16bf*)(uh);  ahh = wmma_bf16(a2, bh2, ahh);

    af += FRAG_ELEMS; hf += FRAG_ELEMS;
    wz += FRAG_ELEMS; wr += FRAG_ELEMS; wh += FRAG_ELEMS;
    uz += FRAG_ELEMS; ur += FRAG_ELEMS; uh += FRAG_ELEMS;
    if (kt + 1 < KT) {      // prefetch next k-tile's weight fragments
      __builtin_prefetch(wz, 0, 3); __builtin_prefetch(wr, 0, 3);
      __builtin_prefetch(wh, 0, 3); __builtin_prefetch(uz, 0, 3);
      __builtin_prefetch(ur, 0, 3); __builtin_prefetch(uh, 0, 3);
    }
  }

  // Gating. C/D layout: acc[v] of lane L -> row = mt*16 + v + 8*(L>>4),
  //                                        col = nc*16 + (L&15).
  const int u    = nc * 16 + (lane & 15);
  const int rowb = mt * 16 + ((lane >> 4) << 3);
  const float bz  = bi[u]       + br[u];
  const float brr = bi[U + u]   + br[U + u];
  const float bxh = bi[2*U + u];
  const float bhh = br[2*U + u];

  // A-fragment coordinates for h_new (k dimension = u):
  //   frag = mt*KT + (u>>5); lane' = (b&15) + 16*((u>>3)&1);
  //   elem = (u&7) | (((u>>4)&1)<<3)
  const int kt2   = u >> 5;
  const int halfu = (u >> 3) & 1;
  const int elem  = (u & 7) | (((u >> 4) & 1) << 3);
  __bf16* fo = Hfragout + (size_t)(mt * KT + kt2) * FRAG_ELEMS
             + (size_t)(16 * halfu) * 16 + elem;

#pragma unroll
  for (int v = 0; v < 8; ++v) {
    const int b = rowb + v;
    float z  = 1.0f / (1.0f + __expf(-(az[v] + bz)));
    float r  = 1.0f / (1.0f + __expf(-(ar[v] + brr)));
    float hc = tanhf(axh[v] + bxh + r * (ahh[v] + bhh));
    float hp = H[(size_t)b * U + u];
    float hn = z * hp + (1.0f - z) * hc;
    Hout[(size_t)b * U + u] = hn;
    fo[(size_t)(b & 15) * 16] = (__bf16)hn;        // next-step bf16 fragment
    if (Yout) Yout[(size_t)b * Ystride + u] = hn;
  }
}

__global__ void copy_final(const float* __restrict__ h0,
                           const float* __restrict__ h1,
                           float* __restrict__ dst) {
  int i = blockIdx.x * blockDim.x + threadIdx.x;
  if (i < Bsz * U) {
    dst[i]           = h0[i];
    dst[Bsz * U + i] = h1[i];
  }
}

extern "C" void kernel_launch(void* const* d_in, const int* in_sizes, int n_in,
                              void* d_out, int out_size, void* d_ws, size_t ws_size,
                              hipStream_t stream) {
  const float* x   = (const float*)d_in[0];
  const float* h0i = (const float*)d_in[1];
  const float* h1i = (const float*)d_in[2];
  const float* W0  = (const float*)d_in[3];
  const float* U0  = (const float*)d_in[4];
  const float* bi0 = (const float*)d_in[5];
  const float* br0 = (const float*)d_in[6];
  const float* W1  = (const float*)d_in[7];
  const float* U1  = (const float*)d_in[8];
  const float* bi1 = (const float*)d_in[9];
  const float* br1 = (const float*)d_in[10];
  float* out = (float*)d_out;

  // Workspace layout
  const size_t WELEMS = (size_t)D * N3;            // bf16 elems per weight matrix
  char* ws = (char*)d_ws;
  __bf16* W0f = (__bf16*)(ws);
  __bf16* U0f = W0f + WELEMS;
  __bf16* W1f = U0f + WELEMS;
  __bf16* U1f = W1f + WELEMS;
  __bf16* xf  = U1f + WELEMS;                      // x fragments, all T (32 MB)
  __bf16* hfr = xf + (size_t)T * AOP_ELEMS;        // 4 x h fragment buffers
  __bf16* h0frag[2] = { hfr,                 hfr + AOP_ELEMS };
  __bf16* h1frag[2] = { hfr + 2 * AOP_ELEMS, hfr + 3 * AOP_ELEMS };
  float*  st  = (float*)(hfr + 4 * AOP_ELEMS);
  float* h0buf[2] = { st,                st +     Bsz * U };
  float* h1buf[2] = { st + 2 * Bsz * U,  st + 3 * Bsz * U };

  // 1) Swizzle all four weight matrices to bf16 WMMA-B fragments (24 MB, L2-hot)
  const int swThreads = NT_ALL * KT * 32;          // 196,608
  swizzle_weights<<<dim3(swThreads / 256), 256, 0, stream>>>(W0, W0f);
  swizzle_weights<<<dim3(swThreads / 256), 256, 0, stream>>>(U0, U0f);
  swizzle_weights<<<dim3(swThreads / 256), 256, 0, stream>>>(W1, W1f);
  swizzle_weights<<<dim3(swThreads / 256), 256, 0, stream>>>(U1, U1f);

  // 2) Pack x for all timesteps into bf16 A-fragments (fully parallel)
  //    operand t: rows b at x + t*D + b*(T*D)
  pack_a_frags<<<dim3(AFRAGS * 32 / 256, T), 256, 0, stream>>>(
      x, (long)T * D, (long)D, xf);

  // 3) Initialize ping-pong recurrent state: f32 copies + bf16 fragments
  hipMemcpyAsync(h0buf[0], h0i, (size_t)Bsz * U * sizeof(float),
                 hipMemcpyDeviceToDevice, stream);
  hipMemcpyAsync(h1buf[0], h1i, (size_t)Bsz * U * sizeof(float),
                 hipMemcpyDeviceToDevice, stream);
  pack_a_frags<<<dim3(AFRAGS * 32 / 256, 1), 256, 0, stream>>>(
      h0i, (long)U, 0, h0frag[0]);
  pack_a_frags<<<dim3(AFRAGS * 32 / 256, 1), 256, 0, stream>>>(
      h1i, (long)U, 0, h1frag[0]);

  // 4) Sequential scan: 2 fused layer-step kernels per timestep
  for (int t = 0; t < T; ++t) {
    const int cur = t & 1, nxt = (t + 1) & 1;
    // Layer 0: A = packed x_t fragments; recurrent h0
    gru_step<<<128, 32, 0, stream>>>(
        xf + (size_t)t * AOP_ELEMS,
        h0buf[cur], h0frag[cur], W0f, U0f, bi0, br0,
        h0buf[nxt], h0frag[nxt], nullptr, 0);
    // Layer 1: A = new h0 fragments; recurrent h1; writes y[:, t, :]
    gru_step<<<128, 32, 0, stream>>>(
        h0frag[nxt],
        h1buf[cur], h1frag[cur], W1f, U1f, bi1, br1,
        h1buf[nxt], h1frag[nxt], out + (size_t)t * U, (long)T * U);
  }

  // 5) Final states (T even -> final state lives in buffer 0)
  copy_final<<<(Bsz * U + 255) / 256, 256, 0, stream>>>(
      h0buf[0], h1buf[0], out + (size_t)Bsz * T * U);
}
